// DLRMambaBackbone_88983132438830
// MI455X (gfx1250) — compile-verified, matrix operand order checked by reading, not run on hardware
//
#include <hip/hip_runtime.h>
#include <hip/hip_bf16.h>
#include <math.h>

// ---------------------------------------------------------------------------
// Types / helpers
// ---------------------------------------------------------------------------
typedef __attribute__((ext_vector_type(16))) __bf16 v16bf;
typedef __attribute__((ext_vector_type(8)))  float  v8f;

#define DEV __device__ __forceinline__

DEV unsigned short f2bf_bits(float f) {
  unsigned u = __builtin_bit_cast(unsigned, f);
  unsigned r = u + 0x7FFFu + ((u >> 16) & 1u);   // round-to-nearest-even
  return (unsigned short)(r >> 16);
}
DEV __bf16 f2bf(float f) {
  unsigned short h = f2bf_bits(f);
  return __builtin_bit_cast(__bf16, h);
}

DEV v8f wmma_bf16(v16bf a, v16bf b, v8f c) {
  // D = A(16x32) * B(32x16) + C(16x16), fp32 accumulate
  return __builtin_amdgcn_wmma_f32_16x16x32_bf16(false, a, false, b,
                                                 (short)0, c, false, false);
}

// async copy 16B global -> LDS (GV mode), tracked by ASYNCcnt
DEV void async_g2l_b128(unsigned lds_byte_addr, const void* gaddr) {
  asm volatile("global_load_async_to_lds_b128 %0, %1, off"
               :: "v"(lds_byte_addr), "v"(gaddr) : "memory");
}
DEV void wait_async0() { asm volatile("s_wait_asynccnt 0" ::: "memory"); }

// fast activations (hw transcendentals, no precise-division expansion)
DEV float silu_fast(float y) {
  return y * __builtin_amdgcn_rcpf(1.f + __expf(-y));
}
DEV float softplus_fast(float x) {
  return fmaxf(x, 0.f) + __logf(1.f + __expf(-fabsf(x)));
}

// ---------------------------------------------------------------------------
// Problem constants
// ---------------------------------------------------------------------------
constexpr int BN   = 8;
constexpr int CIN  = 32;
constexpr int H0   = 112, W0 = 112;
constexpr int E    = 64;
constexpr int HH   = 56,  WW = 56;
constexpr int LL   = HH * WW;          // 3136 tokens
constexpr int RR   = 32;
constexpr int NN   = 64;
constexpr int NBLK = 3;
constexpr int C4   = 128, H4 = 28, W4 = 28;
constexpr int C5   = 256, H5 = 14, W5 = 14;
constexpr int TCH  = 64;               // scan chunk length
constexpr int NCH  = LL / TCH;         // 49 chunks

// d_out float offsets
constexpr size_t OFF_P3 = 0;
constexpr size_t OFF_P4 = (size_t)BN * E  * LL;
constexpr size_t OFF_P5 = OFF_P4 + (size_t)BN * C4 * H4 * W4;
constexpr size_t OFF_ST = OFF_P5 + (size_t)BN * C5 * H5 * W5;

// workspace byte offsets
constexpr size_t WS_X  = 0;
constexpr size_t WS_P  = WS_X  + (size_t)BN * E * LL * 4;        // activations
constexpr size_t WS_Y  = WS_P  + (size_t)BN * LL * 192 * 4;      // proj buffer
constexpr size_t WS_PA = WS_Y  + (size_t)BN * LL * RR * 4;       // y buffer
constexpr size_t WS_PB = WS_PA + (size_t)BN * NCH * RR * NN * 4;
constexpr size_t WS_HI = WS_PB + (size_t)BN * NCH * RR * NN * 4;
constexpr size_t WS_WS = WS_HI + (size_t)BN * NCH * RR * NN * 4; // stem w (bf16)
constexpr size_t WS_W4 = WS_WS + (size_t)9 * E  * CIN * 2;
constexpr size_t WS_W5 = WS_W4 + (size_t)9 * C4 * E   * 2;
constexpr size_t WS_WP = WS_W5 + (size_t)9 * C5 * C4  * 2;
constexpr size_t WS_WO = WS_WP + (size_t)NBLK * 192 * 64 * 2;

// ---------------------------------------------------------------------------
// Weight packing (fp32 -> bf16, WMMA-friendly layouts)
// ---------------------------------------------------------------------------
__global__ void pack_conv_w(const float* __restrict__ w, __bf16* __restrict__ dst,
                            int Cout, int Cin) {
  int i = blockIdx.x * 256 + threadIdx.x;
  int total = 9 * Cout * Cin;
  if (i >= total) return;
  int ci  = i % Cin;
  int t2  = i / Cin;
  int co  = t2 % Cout;
  int tap = t2 / Cout;
  int kh = tap / 3, kw = tap % 3;
  // dst layout: [tap][co][ci]  (contiguous ci per lane column)
  dst[i] = f2bf(w[(((size_t)co * Cin + ci) * 3 + kh) * 3 + kw]);
}

__global__ void pack_proj_w(const float* __restrict__ in_w,
                            const float* __restrict__ dt_w,
                            const float* __restrict__ B_w,
                            const float* __restrict__ C_w,
                            __bf16* __restrict__ dst) {
  int i = blockIdx.x * 256 + threadIdx.x;
  int total = NBLK * 192 * 64;
  if (i >= total) return;
  int blk = i / (192 * 64);
  int rem = i % (192 * 64);
  int n = rem / 64, c = rem % 64;
  float v;
  if      (n < 32)  v = in_w[((size_t)blk * RR + n) * E + c];
  else if (n < 64)  v = dt_w[((size_t)blk * RR + (n - 32)) * E + c];
  else if (n < 128) v = B_w [((size_t)blk * NN + (n - 64)) * E + c];
  else              v = C_w [((size_t)blk * NN + (n - 128)) * E + c];
  dst[i] = f2bf(v);                                  // layout [blk][192][64]
}

__global__ void pack_out_w(const float* __restrict__ ow, __bf16* __restrict__ dst) {
  int i = blockIdx.x * 256 + threadIdx.x;
  if (i >= NBLK * E * RR) return;
  dst[i] = f2bf(ow[i]);                              // layout [blk][64][32]
}

// ---------------------------------------------------------------------------
// 3x3 stride-2 conv + BN + SiLU via implicit GEMM (tap-split, WMMA bf16)
// Fully templated geometry: div/mod by constants, tap loop kept rolled (unroll 1).
// block = 128 (4 waves): 16 output pixels x 64 output channels per workgroup
// ---------------------------------------------------------------------------
template<int Cin, int Cout, int Hi, int Wi, int Ho, int Wo>
__global__ void __launch_bounds__(128)
conv3x3s2_bn_silu(const float* __restrict__ src, const __bf16* __restrict__ wp,
                  const float* __restrict__ g,  const float* __restrict__ bb,
                  const float* __restrict__ mm, const float* __restrict__ vv,
                  float* __restrict__ dst) {
  constexpr int HoWo = Ho * Wo, HiWi = Hi * Wi;
  constexpr int tilesPerImg = (HoWo + 15) / 16;
  constexpr int kchunks = Cin / 32;
  const int b     = blockIdx.x / tilesPerImg;
  const int tile0 = (blockIdx.x % tilesPerImg) * 16;
  const int wave  = threadIdx.x >> 5, lane = threadIdx.x & 31;

  __shared__ __align__(32) __bf16 a_lds[16][32];

  v8f acc = {};

  // staging geometry: thread covers (m = tid>>3, ci0 = (tid&7)*4) -> one
  // 8-byte LDS store of 4 packed bf16 per (tap, kc) iteration
  const int sm   = threadIdx.x >> 3;
  const int sci0 = (threadIdx.x & 7) * 4;
  const int sopix = tile0 + sm;
  const int soh = sopix / Wo, sow = sopix % Wo;      // constant divisors

#pragma unroll 1
  for (int tap = 0; tap < 9; ++tap) {
    const int kh = tap / 3, kw = tap % 3;
    const int ih = soh * 2 - 1 + kh, iw = sow * 2 - 1 + kw;
    const bool inb = (sopix < HoWo) & (ih >= 0) & (ih < Hi) & (iw >= 0) & (iw < Wi);
    for (int kc = 0; kc < kchunks; ++kc) {
      __syncthreads();
      {
        float v0 = 0.f, v1 = 0.f, v2 = 0.f, v3 = 0.f;
        if (inb) {
          const float* s = src + ((size_t)(b * Cin + kc * 32 + sci0)) * HiWi
                               + ih * Wi + iw;
          v0 = s[0]; v1 = s[HiWi]; v2 = s[2 * HiWi]; v3 = s[3 * HiWi];
        }
        unsigned lo = (unsigned)f2bf_bits(v0) | ((unsigned)f2bf_bits(v1) << 16);
        unsigned hi = (unsigned)f2bf_bits(v2) | ((unsigned)f2bf_bits(v3) << 16);
        *(uint2*)&a_lds[sm][sci0] = make_uint2(lo, hi);
      }
      __syncthreads();

      const int row = lane & 15, kb = (lane >> 4) * 16;
      v16bf a = *(const v16bf*)&a_lds[row][kb];
      const int co = blockIdx.y * 64 + wave * 16 + (lane & 15);
      const __bf16* bp = wp + ((size_t)(tap * Cout + co)) * Cin + kc * 32 + kb;
      v16bf bv = *(const v16bf*)bp;
      acc = wmma_bf16(a, bv, acc);
    }
  }

  // fused BN + SiLU epilogue; D layout: lane holds col n, rows r / 8+r
  const int co    = blockIdx.y * 64 + wave * 16 + (lane & 15);
  const float scl = g[co] * rsqrtf(vv[co] + 1e-5f);
  const float bet = bb[co], mean = mm[co];
  const int   mb  = (lane >> 4) * 8;
  float* dp = dst + ((size_t)(b * Cout + co)) * HoWo + tile0 + mb;
#pragma unroll
  for (int r2 = 0; r2 < 8; ++r2) {
    int opix = tile0 + mb + r2;
    if (opix < HoWo) {
      float y = (acc[r2] - mean) * scl + bet;
      dp[r2] = silu_fast(y);
    }
  }
}

// ---------------------------------------------------------------------------
// SS2D fused projection: P[b][l][0:32]=u, [32:64]=softplus(dt+bias),
// [64:128]=Bt, [128:192]=Ct.   K=64 -> 2 WMMAs per 16x16 tile.
// block = 128 (4 waves = 64 outputs), grid.y = 3 output groups
// ---------------------------------------------------------------------------
__global__ void __launch_bounds__(128)
ss2d_proj(const float* __restrict__ X, const __bf16* __restrict__ wp,
          const float* __restrict__ dtb, float* __restrict__ P) {
  constexpr int TL = LL / 16;                  // 196 token tiles per image
  const int b  = blockIdx.x / TL;
  const int l0 = (blockIdx.x % TL) * 16;
  const int wave = threadIdx.x >> 5, lane = threadIdx.x & 31;

  __shared__ __align__(32) __bf16 a_lds[16][64];

  {  // stage xt tile [16 tokens][64 ch] from NCHW (coalesced float4 reads)
    int c = threadIdx.x >> 1, half = threadIdx.x & 1;
    const float* sp = X + ((size_t)(b * E + c)) * LL + l0 + half * 8;
    float4 v0 = *(const float4*)sp;
    float4 v1 = *(const float4*)(sp + 4);
    int m = half * 8;
    a_lds[m + 0][c] = f2bf(v0.x); a_lds[m + 1][c] = f2bf(v0.y);
    a_lds[m + 2][c] = f2bf(v0.z); a_lds[m + 3][c] = f2bf(v0.w);
    a_lds[m + 4][c] = f2bf(v1.x); a_lds[m + 5][c] = f2bf(v1.y);
    a_lds[m + 6][c] = f2bf(v1.z); a_lds[m + 7][c] = f2bf(v1.w);
  }
  __syncthreads();

  const int row = lane & 15, ko = (lane >> 4) * 16;
  v16bf a0 = *(const v16bf*)&a_lds[row][ko];
  v16bf a1 = *(const v16bf*)&a_lds[row][32 + ko];

  const int n = blockIdx.y * 64 + wave * 16 + (lane & 15);
  const __bf16* bp = wp + (size_t)n * 64;
  v16bf b0 = *(const v16bf*)(bp + ko);
  v16bf b1 = *(const v16bf*)(bp + 32 + ko);

  v8f acc = {};
  acc = wmma_bf16(a0, b0, acc);
  acc = wmma_bf16(a1, b1, acc);

  const bool  isdt = (n >= 32 && n < 64);
  const float bias = isdt ? dtb[n - 32] : 0.f;
  const int   mb   = (lane >> 4) * 8;
#pragma unroll
  for (int r2 = 0; r2 < 8; ++r2) {
    float v = acc[r2];
    if (isdt) v = softplus_fast(v + bias);
    P[((size_t)(b * LL + l0 + mb + r2)) * 192 + n] = v;
  }
}

// ---------------------------------------------------------------------------
// Chunked parallel scan. block = 1024 (32 waves); wave=r, lane holds n0=2*lane
// Whole chunk (64 x 192 fp32 = 48KB) is async-DMA'd to LDS before the serial
// recurrence runs, so the dependent loop reads LDS, not HBM/L2.
// ---------------------------------------------------------------------------
DEV void stage_chunk_to_lds(float (*sP)[192], const float* gbase, int tid) {
  unsigned ldsbase = (unsigned)(unsigned long long)(&sP[0][0]);
  const char* g = (const char*)gbase;
#pragma unroll
  for (int i = 0; i < 3; ++i) {                    // 3072 x 16B = 48KB
    unsigned o = (unsigned)(tid + i * 1024) * 16u;
    async_g2l_b128(ldsbase + o, g + o);
  }
  wait_async0();
  __syncthreads();
}

__global__ void __launch_bounds__(1024)
scan_pass1(const float* __restrict__ P, const float* __restrict__ Alog,
           float* __restrict__ Pa, float* __restrict__ Pb) {
  const int chunk = blockIdx.x, b = blockIdx.y;
  const int r = threadIdx.x >> 5, lane = threadIdx.x & 31, n0 = lane * 2;

  __shared__ __align__(16) float sP[TCH][192];
  stage_chunk_to_lds(sP, P + ((size_t)(b * LL + chunk * TCH)) * 192,
                     threadIdx.x);

  const float a0 = -__expf(Alog[r * NN + n0]);
  const float a1 = -__expf(Alog[r * NN + n0 + 1]);
  float aP0 = 1.f, aP1 = 1.f, bP0 = 0.f, bP1 = 0.f;
  for (int t = 0; t < TCH; ++t) {
    float u  = sP[t][r];
    float dt = sP[t][32 + r];
    float2 Bv = *(const float2*)&sP[t][64 + n0];
    float dA0 = __expf(dt * a0), dA1 = __expf(dt * a1);
    float db0 = dt * Bv.x * u,  db1 = dt * Bv.y * u;
    aP0 *= dA0; bP0 = dA0 * bP0 + db0;
    aP1 *= dA1; bP1 = dA1 * bP1 + db1;
  }
  size_t o = ((size_t)(b * NCH + chunk)) * (RR * NN) + r * NN + n0;
  *(float2*)(Pa + o) = make_float2(aP0, aP1);
  *(float2*)(Pb + o) = make_float2(bP0, bP1);
}

__global__ void __launch_bounds__(1024)
scan_pass2(const float* __restrict__ Pa, const float* __restrict__ Pb,
           float* __restrict__ Hinit) {
  const int b = blockIdx.x;
  const int r = threadIdx.x >> 5, lane = threadIdx.x & 31, n0 = lane * 2;
  float h0 = 0.f, h1 = 0.f;
  for (int chunk = 0; chunk < NCH; ++chunk) {
    size_t o = ((size_t)(b * NCH + chunk)) * (RR * NN) + r * NN + n0;
    *(float2*)(Hinit + o) = make_float2(h0, h1);     // state at chunk start
    float2 pa = *(const float2*)(Pa + o);
    float2 pb = *(const float2*)(Pb + o);
    h0 = pa.x * h0 + pb.x;
    h1 = pa.y * h1 + pb.y;
  }
}

__global__ void __launch_bounds__(1024)
scan_pass3(const float* __restrict__ P, const float* __restrict__ Alog,
           const float* __restrict__ Dv, const float* __restrict__ Hinit,
           float* __restrict__ states, float* __restrict__ ybuf) {
  const int chunk = blockIdx.x, b = blockIdx.y;
  const int r = threadIdx.x >> 5, lane = threadIdx.x & 31, n0 = lane * 2;

  __shared__ __align__(16) float sP[TCH][192];
  stage_chunk_to_lds(sP, P + ((size_t)(b * LL + chunk * TCH)) * 192,
                     threadIdx.x);

  const float a0 = -__expf(Alog[r * NN + n0]);
  const float a1 = -__expf(Alog[r * NN + n0 + 1]);
  const float Dr = Dv[r];
  size_t ho = ((size_t)(b * NCH + chunk)) * (RR * NN) + r * NN + n0;
  float2 h = *(const float2*)(Hinit + ho);
  float h0 = h.x, h1 = h.y;
  for (int t = 0; t < TCH; ++t) {
    const size_t l = (size_t)chunk * TCH + t;
    float u  = sP[t][r];
    float dt = sP[t][32 + r];
    float2 Bv = *(const float2*)&sP[t][64 + n0];
    float2 Cv = *(const float2*)&sP[t][128 + n0];
    float dA0 = __expf(dt * a0), dA1 = __expf(dt * a1);
    h0 = dA0 * h0 + dt * Bv.x * u;
    h1 = dA1 * h1 + dt * Bv.y * u;
    size_t so = (((size_t)b * LL + l) * RR + r) * NN + n0;
    *(float2*)(states + so) = make_float2(h0, h1);   // 617 MB stream: coalesced
    float part = h0 * Cv.x + h1 * Cv.y;
#pragma unroll
    for (int off = 16; off > 0; off >>= 1) part += __shfl_xor(part, off, 32);
    if (lane == 0)
      ybuf[((size_t)b * LL + l) * RR + r] = part + Dr * u;  // y + D*u
  }
}

// ---------------------------------------------------------------------------
// Output projection + residual: x += y' @ out_w^T.  K=32 -> one WMMA per tile
// ---------------------------------------------------------------------------
__global__ void __launch_bounds__(128)
ss2d_outproj(const float* __restrict__ ybuf, const __bf16* __restrict__ wo,
             const float* __restrict__ srcX, float* __restrict__ dstX) {
  constexpr int TL = LL / 16;
  const int b  = blockIdx.x / TL;
  const int l0 = (blockIdx.x % TL) * 16;
  const int wave = threadIdx.x >> 5, lane = threadIdx.x & 31;

  __shared__ __align__(32) __bf16 a_lds[16][32];
  {  // thread covers (m = tid>>3, rr0 = (tid&7)*4): float4 load + 8B LDS store
    int m = threadIdx.x >> 3, rr0 = (threadIdx.x & 7) * 4;
    float4 v = *(const float4*)&ybuf[((size_t)(b * LL + l0 + m)) * RR + rr0];
    unsigned lo = (unsigned)f2bf_bits(v.x) | ((unsigned)f2bf_bits(v.y) << 16);
    unsigned hi = (unsigned)f2bf_bits(v.z) | ((unsigned)f2bf_bits(v.w) << 16);
    *(uint2*)&a_lds[m][rr0] = make_uint2(lo, hi);
  }
  __syncthreads();

  const int row = lane & 15, ko = (lane >> 4) * 16;
  v16bf a = *(const v16bf*)&a_lds[row][ko];
  const int c = wave * 16 + (lane & 15);
  v16bf bv = *(const v16bf*)(wo + (size_t)c * RR + ko);
  v8f acc = {};
  acc = wmma_bf16(a, bv, acc);

  const int mb = (lane >> 4) * 8;
  const float* sp = srcX + ((size_t)(b * E + c)) * LL + l0 + mb;
  float*       dp = dstX + ((size_t)(b * E + c)) * LL + l0 + mb;
#pragma unroll
  for (int r2 = 0; r2 < 8; ++r2) dp[r2] = sp[r2] + acc[r2];
}

// ---------------------------------------------------------------------------
// Launch
// ---------------------------------------------------------------------------
extern "C" void kernel_launch(void* const* d_in, const int* in_sizes, int n_in,
                              void* d_out, int out_size, void* d_ws, size_t ws_size,
                              hipStream_t stream) {
  (void)in_sizes; (void)n_in; (void)out_size; (void)ws_size;
  const float* x       = (const float*)d_in[0];
  const float* stem_w  = (const float*)d_in[1];
  const float* stem_g  = (const float*)d_in[2];
  const float* stem_b  = (const float*)d_in[3];
  const float* stem_m  = (const float*)d_in[4];
  const float* stem_v  = (const float*)d_in[5];
  const float* bin_w   = (const float*)d_in[6];
  const float* bdt_w   = (const float*)d_in[7];
  const float* bdt_b   = (const float*)d_in[8];
  const float* bB_w    = (const float*)d_in[9];
  const float* bC_w    = (const float*)d_in[10];
  const float* bA_log  = (const float*)d_in[11];
  const float* bD      = (const float*)d_in[12];
  const float* bout_w  = (const float*)d_in[13];
  const float* p4_w    = (const float*)d_in[14];
  const float* p4_g    = (const float*)d_in[15];
  const float* p4_b    = (const float*)d_in[16];
  const float* p4_m    = (const float*)d_in[17];
  const float* p4_v    = (const float*)d_in[18];
  const float* p5_w    = (const float*)d_in[19];
  const float* p5_g    = (const float*)d_in[20];
  const float* p5_b    = (const float*)d_in[21];
  const float* p5_m    = (const float*)d_in[22];
  const float* p5_v    = (const float*)d_in[23];

  char* ws = (char*)d_ws;
  float*  X     = (float*)(ws + WS_X);
  float*  P     = (float*)(ws + WS_P);
  float*  ybuf  = (float*)(ws + WS_Y);
  float*  Pa    = (float*)(ws + WS_PA);
  float*  Pb    = (float*)(ws + WS_PB);
  float*  Hinit = (float*)(ws + WS_HI);
  __bf16* wS    = (__bf16*)(ws + WS_WS);
  __bf16* w4    = (__bf16*)(ws + WS_W4);
  __bf16* w5    = (__bf16*)(ws + WS_W5);
  __bf16* wP    = (__bf16*)(ws + WS_WP);
  __bf16* wO    = (__bf16*)(ws + WS_WO);
  float* outF = (float*)d_out;

  // weight packing (fp32 -> bf16, WMMA layouts)
  pack_conv_w<<<(9 * E  * CIN + 255) / 256, 256, 0, stream>>>(stem_w, wS, E,  CIN);
  pack_conv_w<<<(9 * C4 * E   + 255) / 256, 256, 0, stream>>>(p4_w,   w4, C4, E);
  pack_conv_w<<<(9 * C5 * C4  + 255) / 256, 256, 0, stream>>>(p5_w,   w5, C5, C4);
  pack_proj_w<<<(NBLK * 192 * 64 + 255) / 256, 256, 0, stream>>>(bin_w, bdt_w, bB_w, bC_w, wP);
  pack_out_w <<<(NBLK * E * RR   + 255) / 256, 256, 0, stream>>>(bout_w, wO);

  // stem: 32 -> 64, 112 -> 56  (196 tiles/img)
  conv3x3s2_bn_silu<CIN, E, H0, W0, HH, WW>
      <<<dim3(BN * 196, 1), 128, 0, stream>>>(
      x, wS, stem_g, stem_b, stem_m, stem_v, X);

  for (int blk = 0; blk < NBLK; ++blk) {
    const float* Alog = bA_log + (size_t)blk * RR * NN;
    ss2d_proj<<<dim3(BN * (LL / 16), 3), 128, 0, stream>>>(
        X, wP + (size_t)blk * 192 * 64, bdt_b + (size_t)blk * RR, P);
    scan_pass1<<<dim3(NCH, BN), 1024, 0, stream>>>(P, Alog, Pa, Pb);
    scan_pass2<<<dim3(BN), 1024, 0, stream>>>(Pa, Pb, Hinit);
    float* st = outF + OFF_ST + (size_t)blk * BN * LL * RR * NN;
    scan_pass3<<<dim3(NCH, BN), 1024, 0, stream>>>(
        P, Alog, bD + (size_t)blk * RR, Hinit, st, ybuf);
    float* dstX = (blk == NBLK - 1) ? (outF + OFF_P3) : X;
    ss2d_outproj<<<dim3(BN * (LL / 16)), 128, 0, stream>>>(
        ybuf, wO + (size_t)blk * E * RR, X, dstX);
  }

  // p4: 64 -> 128, 56 -> 28  (49 tiles/img)
  conv3x3s2_bn_silu<E, C4, HH, WW, H4, W4>
      <<<dim3(BN * 49, 2), 128, 0, stream>>>(
      outF + OFF_P3, w4, p4_g, p4_b, p4_m, p4_v, outF + OFF_P4);
  // p5: 128 -> 256, 28 -> 14 (196 px -> 13 tiles with tail masking)
  conv3x3s2_bn_silu<C4, C5, H4, W4, H5, W5>
      <<<dim3(BN * 13, 4), 128, 0, stream>>>(
      outF + OFF_P4, w5, p5_g, p5_b, p5_m, p5_v, outF + OFF_P5);
}
